// MalConv1_57629871177878
// MI455X (gfx1250) — compile-verified, matrix operand order checked by reading, not run on hardware
//
#include <hip/hip_runtime.h>
#include <math.h>

typedef __attribute__((ext_vector_type(16))) __bf16 v16bf;
typedef __attribute__((ext_vector_type(8)))  __bf16 v8bf;
typedef __attribute__((ext_vector_type(8)))  float  v8f;
typedef __attribute__((ext_vector_type(4)))  float  v4f;

#define WINDOW 512
#define E_DIM  8
#define KDIM   4096          // WINDOW * E_DIM, contiguous per window in x[B,L,E]
#define CDIM   128
#define BATCH  8
#define LSEQ   2000000
#define NWIN   (LSEQ / WINDOW)        // 3906
#define MROWS  32                     // rows per workgroup (2 M-tiles per wave)
#define MTILES ((NWIN + MROWS - 1) / MROWS)   // 123

#define KCHUNK 512                    // K staged per LDS round
#define NSTEPS (KCHUNK / 32)          // 16 WMMA K-steps per chunk
// LDS A chunk: [NSTEPS][4 groups][32 rows][8 bf16] = 32 KB
#define ALDS_ELEMS (NSTEPS * 4 * MROWS * 8)

// ---------------------------------------------------------------------------
// Prep: repack conv weights [C][E][W] fp32 -> Bt[C][K] bf16 with k = w*E + e
// ---------------------------------------------------------------------------
__global__ void prep_weights(const float* __restrict__ w1,
                             const float* __restrict__ w2,
                             __bf16* __restrict__ wb1,
                             __bf16* __restrict__ wb2) {
    int idx = blockIdx.x * blockDim.x + threadIdx.x;
    if (idx >= CDIM * KDIM) return;
    int c = idx / KDIM;
    int k = idx % KDIM;
    int e = k % E_DIM;
    int w = k / E_DIM;
    int src = c * (E_DIM * WINDOW) + e * WINDOW + w;
    wb1[idx] = (__bf16)w1[src];
    wb2[idx] = (__bf16)w2[src];
}

__global__ void init_pooled(float* __restrict__ pooled) {
    int i = blockIdx.x * blockDim.x + threadIdx.x;
    if (i < BATCH * CDIM) pooled[i] = -INFINITY;
}

// Sign-aware float atomic max (pooled is initialized to -inf, values finite).
__device__ inline void atomicMaxF(float* addr, float val) {
    if (val >= 0.0f) {
        atomicMax((int*)addr, __float_as_int(val));
    } else {
        atomicMin((unsigned int*)addr, __float_as_uint(val));
    }
}

// ---------------------------------------------------------------------------
// Main kernel: per (batch, 32-row tile) workgroup of 8 waves.
// A chunk (32 rows x 512 K) staged once per workgroup into LDS as bf16 in
// fragment order. Each wave owns a 16-column tile and computes TWO 16-row
// M-tiles against the same B fragments: 4 WMMAs per 8 loads per K-step.
// ---------------------------------------------------------------------------
__global__ __launch_bounds__(256) void malconv_gemm(
    const float*  __restrict__ x,
    const __bf16* __restrict__ wb1,
    const __bf16* __restrict__ wb2,
    const float*  __restrict__ bias1,
    const float*  __restrict__ bias2,
    float*        __restrict__ pooled) {

    __shared__ __bf16 alds[ALDS_ELEMS];

    const int wg    = blockIdx.x;
    const int b     = wg / MTILES;
    const int mtile = wg % MTILES;
    const int m0    = mtile * MROWS;

    const int tid  = threadIdx.x;
    const int lane = tid & 31;
    const int wave = tid >> 5;           // 0..7 -> N tile
    const int lo   = lane & 15;
    const int hi   = lane >> 4;          // K-half selector for fragments
    const int n0   = wave * 16;

    // ---- loader-side constants: thread owns fixed (row, 8-col group) -----
    // gidx = tid + rep*256, rep = 0..7 covers 2048 groups of the 32x512 chunk
    //   row = gidx % 32  (constant = tid & 31)
    //   g   = (gidx/32) % 4 (constant = (tid>>5) & 3)
    //   s   = gidx / 128 = (tid>>7) + 2*rep
    const int l_row = tid & 31;
    const int l_g   = (tid >> 5) & 3;
    const int l_s0  = tid >> 7;                  // 0 or 1
    int ldRow = m0 + l_row;
    if (ldRow >= NWIN) ldRow = NWIN - 1;         // clamp tail-tile loads
    const float* __restrict__ ldrow =
        x + ((size_t)b * NWIN + (size_t)ldRow) * KDIM;

    // ---- compute-side pointers -------------------------------------------
    const int colB = n0 + lo;
    const __bf16* __restrict__ brow1 = wb1 + (size_t)colB * KDIM;
    const __bf16* __restrict__ brow2 = wb2 + (size_t)colB * KDIM;
    const int kb = hi * 8;               // per-lane K sub-offset for B frags

    v8f acc1_0 = {}, acc2_0 = {};        // M-tile 0 (rows m0 .. m0+15)
    v8f acc1_1 = {}, acc2_1 = {};        // M-tile 1 (rows m0+16 .. m0+31)

    for (int kc = 0; kc < KDIM; kc += KCHUNK) {
        // ================= stage A chunk into LDS (bf16) ==================
#pragma unroll
        for (int rep = 0; rep < 8; ++rep) {
            const int s = l_s0 + 2 * rep;                // 0..15
            const float* src = ldrow + kc + s * 32 + l_g * 8;
            v4f f0 = *(const v4f*)(src + 0);
            v4f f1 = *(const v4f*)(src + 4);
            v8bf p;
#pragma unroll
            for (int i = 0; i < 4; ++i) {
                p[i]     = (__bf16)f0[i];
                p[4 + i] = (__bf16)f1[i];
            }
            *(v8bf*)&alds[(((s * 4 + l_g) * MROWS) + l_row) * 8] = p;
        }
        __syncthreads();

        // ================= WMMA over the staged chunk =====================
#pragma unroll 4
        for (int s = 0; s < NSTEPS; ++s) {
            // A fragments for both M-tiles (contiguous 16-B groups in LDS)
            const int base_lo = ((s * 4 + hi)     * MROWS + lo) * 8;
            const int base_hi = ((s * 4 + hi + 2) * MROWS + lo) * 8;
            v8bf a0l = *(const v8bf*)&alds[base_lo];
            v8bf a0h = *(const v8bf*)&alds[base_hi];
            v8bf a1l = *(const v8bf*)&alds[base_lo + 16 * 8];
            v8bf a1h = *(const v8bf*)&alds[base_hi + 16 * 8];
            v16bf af0, af1;
#pragma unroll
            for (int i = 0; i < 8; ++i) {
                af0[i] = a0l[i];  af0[8 + i] = a0h[i];
                af1[i] = a1l[i];  af1[8 + i] = a1h[i];
            }

            // B fragments from global (per-wave column strip, L2 resident)
            const int k0 = kc + s * 32;
            const __bf16* pb1 = brow1 + k0 + kb;
            const __bf16* pb2 = brow2 + k0 + kb;
            v8bf b1a = *(const v8bf*)(pb1 +  0);
            v8bf b1b = *(const v8bf*)(pb1 + 16);
            v8bf b2a = *(const v8bf*)(pb2 +  0);
            v8bf b2b = *(const v8bf*)(pb2 + 16);
            v16bf bfrag1, bfrag2;
#pragma unroll
            for (int i = 0; i < 8; ++i) {
                bfrag1[i] = b1a[i];  bfrag1[8 + i] = b1b[i];
                bfrag2[i] = b2a[i];  bfrag2[8 + i] = b2b[i];
            }

            acc1_0 = __builtin_amdgcn_wmma_f32_16x16x32_bf16(
                false, af0, false, bfrag1, (short)0, acc1_0, false, false);
            acc2_0 = __builtin_amdgcn_wmma_f32_16x16x32_bf16(
                false, af0, false, bfrag2, (short)0, acc2_0, false, false);
            acc1_1 = __builtin_amdgcn_wmma_f32_16x16x32_bf16(
                false, af1, false, bfrag1, (short)0, acc1_1, false, false);
            acc2_1 = __builtin_amdgcn_wmma_f32_16x16x32_bf16(
                false, af1, false, bfrag2, (short)0, acc2_1, false, false);
        }
        __syncthreads();
    }

    // ---- epilogue: bias + sigmoid gate + masked row max ------------------
    const int c  = n0 + lo;              // output column for this lane
    const float bv = bias1[c];
    const float bg = bias2[c];

    float mx = -INFINITY;
#pragma unroll
    for (int v = 0; v < 8; ++v) {
        // M-tile 0: rows m0 + v + 8*hi
        {
            int mrow   = m0 + v + 8 * hi;
            float val  = acc1_0[v] + bv;
            float gate = 1.0f / (1.0f + __expf(-(acc2_0[v] + bg)));
            float g    = val * gate;
            if (mrow < NWIN && g > mx) mx = g;
        }
        // M-tile 1: rows m0 + 16 + v + 8*hi
        {
            int mrow   = m0 + 16 + v + 8 * hi;
            float val  = acc1_1[v] + bv;
            float gate = 1.0f / (1.0f + __expf(-(acc2_1[v] + bg)));
            float g    = val * gate;
            if (mrow < NWIN && g > mx) mx = g;
        }
    }
    // combine the two half-wave row groups (same column, xor-16 partner)
    float other = __shfl_xor(mx, 16, 32);
    mx = fmaxf(mx, other);
    if (hi == 0) atomicMaxF(pooled + b * CDIM + c, mx);
}

// ---------------------------------------------------------------------------
// Tiny MLP head: pooled[8][128] -> relu(fc1) -> fc2 -> out[8][2]
// ---------------------------------------------------------------------------
__global__ void head_kernel(const float* __restrict__ pooled,
                            const float* __restrict__ fc1_w,
                            const float* __restrict__ fc1_b,
                            const float* __restrict__ fc2_w,
                            const float* __restrict__ fc2_b,
                            float* __restrict__ out) {
    __shared__ float h[CDIM];
    const int c = threadIdx.x;           // 128 threads
    for (int b = 0; b < BATCH; ++b) {
        float acc = fc1_b[c];
        const float* p = pooled + b * CDIM;
#pragma unroll 8
        for (int j = 0; j < CDIM; ++j) acc += p[j] * fc1_w[c * CDIM + j];
        h[c] = fmaxf(acc, 0.0f);
        __syncthreads();
        if (c < 2) {
            float o = fc2_b[c];
#pragma unroll 8
            for (int j = 0; j < CDIM; ++j) o += h[j] * fc2_w[c * CDIM + j];
            out[b * 2 + c] = o;
        }
        __syncthreads();
    }
}

extern "C" void kernel_launch(void* const* d_in, const int* in_sizes, int n_in,
                              void* d_out, int out_size, void* d_ws, size_t ws_size,
                              hipStream_t stream) {
    const float* x   = (const float*)d_in[0];
    const float* c1w = (const float*)d_in[1];
    const float* c1b = (const float*)d_in[2];
    const float* c2w = (const float*)d_in[3];
    const float* c2b = (const float*)d_in[4];
    const float* f1w = (const float*)d_in[5];
    const float* f1b = (const float*)d_in[6];
    const float* f2w = (const float*)d_in[7];
    const float* f2b = (const float*)d_in[8];
    float* out = (float*)d_out;

    // Workspace layout: wb1 (1 MB bf16) | wb2 (1 MB bf16) | pooled (4 KB f32)
    char* ws = (char*)d_ws;
    __bf16* wb1    = (__bf16*)ws;
    __bf16* wb2    = (__bf16*)(ws + (size_t)CDIM * KDIM * 2);
    float*  pooled = (float*)(ws + (size_t)CDIM * KDIM * 4);

    prep_weights<<<(CDIM * KDIM + 255) / 256, 256, 0, stream>>>(c1w, c2w, wb1, wb2);
    init_pooled<<<(BATCH * CDIM + 255) / 256, 256, 0, stream>>>(pooled);
    malconv_gemm<<<BATCH * MTILES, 256, 0, stream>>>(x, wb1, wb2, c1b, c2b, pooled);
    head_kernel<<<1, CDIM, 0, stream>>>(pooled, f1w, f1b, f2w, f2b, out);
}
